// GraphSAGE_47364899340883
// MI455X (gfx1250) — matrix operand order repaired and checked by reading, not compiled, not run on hardware
//
#include <hip/hip_runtime.h>
#include <math.h>
#include <stdint.h>

// ---------------------------------------------------------------------------
// Types for CDNA5 WMMA
// ---------------------------------------------------------------------------
typedef __attribute__((ext_vector_type(16))) __bf16 v16bf;
typedef __attribute__((ext_vector_type(8)))  __bf16 v8bf;
typedef __attribute__((ext_vector_type(8)))  float  v8f;
typedef __attribute__((ext_vector_type(4)))  int    v4i;

__device__ __forceinline__ __bf16 f2bf(float f) { return (__bf16)f; }

// gfx1250 async global->LDS copy availability (device pass only)
#if defined(__HIP_DEVICE_COMPILE__)
#if __has_builtin(__builtin_amdgcn_global_load_async_to_lds_b128)
#define HAVE_ASYNC_LDS 1
#endif
#endif

// ---------------------------------------------------------------------------
// Weight pre-swizzle: f32 [128K x 128N] row-major  ->  bf16 fragment-major.
// For tile T = kt*8+nt (kt 0..3, nt 0..7), lane (half=lane>>4, m=lane&15):
//   elem j of the lane's WMMA B fragment = W[kt*32 + (j>>3)*16 + half*8 + (j&7)][nt*16 + m]
// Stored as two wave-contiguous 16B chunks per lane:
//   out[T*512 + jhalf*256 + lane*8 + (j&7)]
// so a full-wave fragment load is two ds_load_b128 over contiguous 512B.
// ---------------------------------------------------------------------------
__global__ __launch_bounds__(256) void conv_weight_frag(
    const float* __restrict__ W, __bf16* __restrict__ out)
{
    int u = blockIdx.x * blockDim.x + threadIdx.x;   // 2048 units of 8 elems
    if (u >= 2048) return;
    const int jhalf = u & 1;
    const int lane  = (u >> 1) & 31;
    const int T     = u >> 6;            // 0..31
    const int kt = T >> 3, nt = T & 7;
    const int half = lane >> 4, m = lane & 15;
    const int col  = nt * 16 + m;
    const int krow = kt * 32 + half * 8 + jhalf * 16;
    __bf16* dst = out + T * 512 + jhalf * 256 + lane * 8;
#pragma unroll
    for (int j = 0; j < 8; ++j)
        dst[j] = f2bf(W[(krow + j) * 128 + col]);
}

// ---------------------------------------------------------------------------
// Stage a 32KB fragment-major weight buffer into LDS.
// Uses GLOBAL_LOAD_ASYNC_TO_LDS_B128 when available (ASYNCcnt-tracked DMA,
// no VGPR round trip), else a plain b128 register copy.
// ---------------------------------------------------------------------------
__device__ __forceinline__ void stage_weights_lds(
    const __bf16* __restrict__ gsrc, __bf16* __restrict__ lds, int tid)
{
    const uint4* g = (const uint4*)gsrc;   // 2048 x 16B
    uint4* l = (uint4*)lds;
#if defined(HAVE_ASYNC_LDS)
#pragma unroll 2
    for (int i = tid; i < 2048; i += 256) {
        __builtin_amdgcn_global_load_async_to_lds_b128(
            (__attribute__((address_space(1))) v4i*)(uintptr_t)(g + i),
            (__attribute__((address_space(3))) v4i*)(uint32_t)(uintptr_t)(l + i),
            0, 0);
    }
#if __has_builtin(__builtin_amdgcn_s_wait_asynccnt)
    __builtin_amdgcn_s_wait_asynccnt(0);
#else
    asm volatile("s_wait_asynccnt 0" ::: "memory");
#endif
#else
    for (int i = tid; i < 2048; i += 256) l[i] = g[i];
#endif
}

// ---------------------------------------------------------------------------
// WMMA GEMM:  out[R,128] = epilogue( A0*rs0 @ B0  (+ A1 @ B1) )
//   A: [R,128] f32 row-major (optionally scaled per-row by rs, f32->bf16)
//   B: pre-swizzled bf16 fragment-major (16384 elems), staged to LDS
//   epilogue: y = acc * svec[col] + tvec[col];  optional ReLU
// Block = 256 threads (8 waves) -> 128-row strip; wave = 16-row M-tile x
// 8 N-tiles (8x v8f accumulators).
// ---------------------------------------------------------------------------
__device__ __forceinline__ void gemm_accum_pass(
    const float* __restrict__ A, const float* __restrict__ rowscale,
    const __bf16* __restrict__ sB, int row, bool rowok, int half, int m,
    int lane, v8f acc[8])
{
    const float* arow = rowok ? (A + (long long)row * 128) : nullptr;
    float rs = 1.0f;
    if (rowok && rowscale) rs = rowscale[row];

#pragma unroll
    for (int kt = 0; kt < 4; ++kt) {
        const int k0 = kt * 32;
        v16bf a;
        if (rowok) {
            // ISA 16-bit A 16x32 layout:
            //  lanes 0-15 : elems 0-7 -> K=k0+0..7,   elems 8-15 -> K=k0+16..23
            //  lanes 16-31: elems 0-7 -> K=k0+8..15,  elems 8-15 -> K=k0+24..31
            const float4 p0 = *(const float4*)(arow + k0 + half * 8);
            const float4 p1 = *(const float4*)(arow + k0 + half * 8 + 4);
            const float4 p2 = *(const float4*)(arow + k0 + 16 + half * 8);
            const float4 p3 = *(const float4*)(arow + k0 + 16 + half * 8 + 4);
            a[0]  = f2bf(p0.x * rs); a[1]  = f2bf(p0.y * rs);
            a[2]  = f2bf(p0.z * rs); a[3]  = f2bf(p0.w * rs);
            a[4]  = f2bf(p1.x * rs); a[5]  = f2bf(p1.y * rs);
            a[6]  = f2bf(p1.z * rs); a[7]  = f2bf(p1.w * rs);
            a[8]  = f2bf(p2.x * rs); a[9]  = f2bf(p2.y * rs);
            a[10] = f2bf(p2.z * rs); a[11] = f2bf(p2.w * rs);
            a[12] = f2bf(p3.x * rs); a[13] = f2bf(p3.y * rs);
            a[14] = f2bf(p3.z * rs); a[15] = f2bf(p3.w * rs);
        } else {
#pragma unroll
            for (int j = 0; j < 16; ++j) a[j] = f2bf(0.0f);
        }

#pragma unroll
        for (int nt = 0; nt < 8; ++nt) {
            const int T = kt * 8 + nt;
            // Fragment-major layout: two contiguous 16B chunks per lane
            const v8bf lo = *(const v8bf*)(sB + T * 512 + lane * 8);
            const v8bf hi = *(const v8bf*)(sB + T * 512 + 256 + lane * 8);
            const v16bf b = __builtin_shufflevector(
                lo, hi, 0, 1, 2, 3, 4, 5, 6, 7, 8, 9, 10, 11, 12, 13, 14, 15);
            acc[nt] = __builtin_amdgcn_wmma_f32_16x16x32_bf16(
                false, a, false, b, (short)0, acc[nt], false, false);
        }
    }
}

__global__ __launch_bounds__(256) void gemm128_wmma(
    const float* __restrict__ A0, const float* __restrict__ rs0,
    const __bf16* __restrict__ B0f,
    const float* __restrict__ A1, const __bf16* __restrict__ B1f,
    const float* __restrict__ svec, const float* __restrict__ tvec,
    float* __restrict__ out, int R, int relu)
{
    __shared__ __bf16 sB[128 * 128];   // 32 KB staged weight fragments

    const int tid = threadIdx.x;
    stage_weights_lds(B0f, sB, tid);
    __syncthreads();

    const int wave = tid >> 5;
    const int lane = tid & 31;
    const int half = lane >> 4;
    const int m    = lane & 15;
    const int rowbase = blockIdx.x * 128 + wave * 16;
    const int row  = rowbase + m;
    const bool rowok = row < R;

    v8f acc[8];
#pragma unroll
    for (int nt = 0; nt < 8; ++nt)
#pragma unroll
        for (int r = 0; r < 8; ++r) acc[nt][r] = 0.0f;

    gemm_accum_pass(A0, rs0, sB, row, rowok, half, m, lane, acc);

    if (A1 != nullptr) {
        __syncthreads();                 // all waves done reading B0
        stage_weights_lds(B1f, sB, tid);
        __syncthreads();
        gemm_accum_pass(A1, nullptr, sB, row, rowok, half, m, lane, acc);
    }

    // Epilogue: y = acc*s[col] + t[col] (+ReLU). C/D layout: VGPR r ->
    // M = r + half*8, N = (lane&15) + nt*16.
#pragma unroll
    for (int nt = 0; nt < 8; ++nt) {
        const int col = nt * 16 + m;
        const float s = svec[col];
        const float t = tvec[col];
#pragma unroll
        for (int r = 0; r < 8; ++r) {
            const int orow = rowbase + half * 8 + r;
            if (orow < R) {
                float v = acc[nt][r] * s + t;
                if (relu) v = fmaxf(v, 0.0f);
                out[(long long)orow * 128 + col] = v;
            }
        }
    }
}

// ---------------------------------------------------------------------------
// Helper kernels
// ---------------------------------------------------------------------------
__global__ void zero_f32(float* __restrict__ p, long long n) {
    long long i = (long long)blockIdx.x * blockDim.x + threadIdx.x;
    if (i < n) p[i] = 0.0f;
}

// One wave per edge: gather 128 floats of h[src] as float4/lane, atomically
// accumulate into L2-resident agg[dst]; count degree.
__global__ __launch_bounds__(256) void scatter_edges(
    const float* __restrict__ h, const int* __restrict__ src,
    const int* __restrict__ dst, float* __restrict__ agg,
    float* __restrict__ deg, int E)
{
    long long gid = (long long)blockIdx.x * blockDim.x + threadIdx.x;
    int e    = (int)(gid >> 5);
    int lane = (int)(gid & 31);
    if (e >= E) return;
    const int s = src[e];
    const int d = dst[e];
    const float4 v = *((const float4*)(h + (long long)s * 128) + lane);
    float* ap = agg + (long long)d * 128 + lane * 4;
    atomicAdd(ap + 0, v.x);
    atomicAdd(ap + 1, v.y);
    atomicAdd(ap + 2, v.z);
    atomicAdd(ap + 3, v.w);
    if (lane == 0) atomicAdd(&deg[d], 1.0f);
}

__global__ void recip_clamp(float* __restrict__ p, int n) {
    int i = blockIdx.x * blockDim.x + threadIdx.x;
    if (i < n) p[i] = 1.0f / fmaxf(p[i], 1.0f);
}

// One wave per node: accumulate h[node] into pooled[batch[node]].
__global__ __launch_bounds__(256) void pool_nodes(
    const float* __restrict__ h, const int* __restrict__ batch,
    float* __restrict__ pooled, float* __restrict__ cnt, int N)
{
    long long gid = (long long)blockIdx.x * blockDim.x + threadIdx.x;
    int i    = (int)(gid >> 5);
    int lane = (int)(gid & 31);
    if (i >= N) return;
    const int g = batch[i];
    const float4 v = *((const float4*)(h + (long long)i * 128) + lane);
    float* pp = pooled + (long long)g * 128 + lane * 4;
    atomicAdd(pp + 0, v.x);
    atomicAdd(pp + 1, v.y);
    atomicAdd(pp + 2, v.z);
    atomicAdd(pp + 3, v.w);
    if (lane == 0) atomicAdd(&cnt[g], 1.0f);
}

// Per-column scale/shift for the GEMM epilogue.
// mode 0: s = 1, t = bias
// mode 1: s = g*rsqrt(v+eps), t = (bias - mean)*s + beta   (BN fold)
__global__ void prep_st(
    const float* __restrict__ bias, const float* __restrict__ g,
    const float* __restrict__ beta, const float* __restrict__ mean,
    const float* __restrict__ var, float* __restrict__ svec,
    float* __restrict__ tvec, int mode)
{
    int i = threadIdx.x;
    if (i >= 128) return;
    if (mode == 0) {
        svec[i] = 1.0f;
        tvec[i] = bias[i];
    } else {
        float s = g[i] * rsqrtf(var[i] + 1e-5f);
        svec[i] = s;
        tvec[i] = (bias[i] - mean[i]) * s + beta[i];
    }
}

// Final tiny GEMM: [G,128] @ [128,C] + b   (G*C = 5120 elements)
__global__ void head2_kernel(const float* __restrict__ h1,
                             const float* __restrict__ w2,
                             const float* __restrict__ b2,
                             float* __restrict__ out, int G, int C)
{
    int gid = blockIdx.x * blockDim.x + threadIdx.x;
    if (gid >= G * C) return;
    int g = gid / C, c = gid % C;
    float acc = b2[c];
#pragma unroll 8
    for (int k = 0; k < 128; ++k) acc += h1[g * 128 + k] * w2[k * C + c];
    out[gid] = acc;
}

// ---------------------------------------------------------------------------
// Launch
// ---------------------------------------------------------------------------
extern "C" void kernel_launch(void* const* d_in, const int* in_sizes, int n_in,
                              void* d_out, int out_size, void* d_ws, size_t ws_size,
                              hipStream_t stream)
{
    const int N = in_sizes[2];          // batch vector length = #nodes
    const int E = in_sizes[1] / 2;      // edge_index is [2,E]
    const int G = 512, C = 10;

    const float* x      = (const float*)d_in[0];
    const int*   ei     = (const int*)d_in[1];
    const int*   src    = ei;
    const int*   dst    = ei + E;
    const int*   batch  = (const int*)d_in[2];
    const float* emb_w  = (const float*)d_in[3];
    const float* emb_b  = (const float*)d_in[4];
    const float* lin1_w = (const float*)d_in[26];
    const float* lin1_b = (const float*)d_in[27];
    const float* lin2_w = (const float*)d_in[28];
    const float* lin2_b = (const float*)d_in[29];

    // Workspace layout (floats, then bf16 weight-fragment buffers)
    float* ws     = (float*)d_ws;
    float* h0     = ws;                              // N*128
    float* agg    = h0 + (size_t)N * 128;            // N*128
    float* deg    = agg + (size_t)N * 128;           // N    (contiguous w/ agg)
    float* pooled = deg + N;                         // G*128
    float* cnt    = pooled + (size_t)G * 128;        // G    (contiguous w/ pooled)
    float* h1     = cnt + G;                         // G*128
    float* svec   = h1 + (size_t)G * 128;            // 128
    float* tvec   = svec + 128;                      // 128
    __bf16* wb    = (__bf16*)(tvec + 128);           // 8 x 16384 bf16 fragments
    __bf16* wfrag[8];
    for (int i = 0; i < 8; ++i) wfrag[i] = wb + (size_t)i * 16384;

    // Pre-swizzle all 128x128 weights into WMMA fragment-major bf16
    const float* wsrc[8] = {
        emb_w,
        (const float*)d_in[5 + 0],  (const float*)d_in[5 + 2],        // wl0, wr0
        (const float*)d_in[5 + 7],  (const float*)d_in[5 + 9],        // wl1, wr1
        (const float*)d_in[5 + 14], (const float*)d_in[5 + 16],       // wl2, wr2
        lin1_w
    };
    for (int i = 0; i < 8; ++i)
        conv_weight_frag<<<8, 256, 0, stream>>>(wsrc[i], wfrag[i]);

    // Output layout: logits [G*C], then middle0/1/2 [N*128] each
    float* outv = (float*)d_out;
    float* mid0 = outv + (size_t)G * C;
    float* mid1 = mid0 + (size_t)N * 128;
    float* mid2 = mid1 + (size_t)N * 128;
    float* mid[3] = {mid0, mid1, mid2};

    const int gemmBlocksN = (N + 127) / 128;

    // ---- Embedding: h0 = x @ emb_w + emb_b ----
    prep_st<<<1, 128, 0, stream>>>(emb_b, nullptr, nullptr, nullptr, nullptr,
                                   svec, tvec, 0);
    gemm128_wmma<<<gemmBlocksN, 256, 0, stream>>>(
        x, nullptr, wfrag[0], nullptr, nullptr, svec, tvec, h0, N, 0);

    // ---- 3 SAGE layers ----
    const float* hprev = h0;
    for (int l = 0; l < 3; ++l) {
        const float* bl = (const float*)d_in[5 + 7 * l + 1];
        const float* bg = (const float*)d_in[5 + 7 * l + 3];
        const float* bb = (const float*)d_in[5 + 7 * l + 4];
        const float* bm = (const float*)d_in[5 + 7 * l + 5];
        const float* bv = (const float*)d_in[5 + 7 * l + 6];

        const long long nz = (long long)N * 128 + N;       // agg + deg
        zero_f32<<<(int)((nz + 255) / 256), 256, 0, stream>>>(agg, nz);

        const long long tE = (long long)E * 32;            // wave per edge
        scatter_edges<<<(int)((tE + 255) / 256), 256, 0, stream>>>(
            hprev, src, dst, agg, deg, E);

        recip_clamp<<<(N + 255) / 256, 256, 0, stream>>>(deg, N);

        prep_st<<<1, 128, 0, stream>>>(bl, bg, bb, bm, bv, svec, tvec, 1);
        gemm128_wmma<<<gemmBlocksN, 256, 0, stream>>>(
            agg, deg, wfrag[1 + 2 * l], hprev, wfrag[2 + 2 * l],
            svec, tvec, mid[l], N, 1);
        hprev = mid[l];
    }

    // ---- Global mean pool ----
    const long long pz = (long long)G * 128 + G;           // pooled + cnt
    zero_f32<<<(int)((pz + 255) / 256), 256, 0, stream>>>(pooled, pz);
    const long long tN = (long long)N * 32;                // wave per node
    pool_nodes<<<(int)((tN + 255) / 256), 256, 0, stream>>>(
        hprev, batch, pooled, cnt, N);
    recip_clamp<<<(G + 255) / 256, 256, 0, stream>>>(cnt, G);

    // ---- Head: relu(pooled/cnt @ lin1 + b1) then @ lin2 + b2 ----
    prep_st<<<1, 128, 0, stream>>>(lin1_b, nullptr, nullptr, nullptr, nullptr,
                                   svec, tvec, 0);
    gemm128_wmma<<<(G + 127) / 128, 256, 0, stream>>>(
        pooled, cnt, wfrag[7], nullptr, nullptr, svec, tvec, h1, G, 1);

    head2_kernel<<<(G * C + 255) / 256, 256, 0, stream>>>(
        h1, lin2_w, lin2_b, outv, G, C);
}
